// MultiHeadAttention_91233695302214
// MI455X (gfx1250) — matrix operand order repaired and checked by reading, not compiled, not run on hardware
//
#include <hip/hip_runtime.h>

typedef __attribute__((ext_vector_type(16))) __bf16        v16bf;
typedef __attribute__((ext_vector_type(8)))  float         v8f;
typedef __attribute__((ext_vector_type(4)))  unsigned int  v4u;

#define NS 8
#define LL 2048
#define II 1024
#define OO 1024
#define NROWS (NS * LL)

// ---- fp32 -> bf16 split helpers (round-to-nearest-even for hi) ----
__device__ __forceinline__ unsigned short f2bf(float x) {
    unsigned int u = __float_as_uint(x);
    u += 0x7FFFu + ((u >> 16) & 1u);
    return (unsigned short)(u >> 16);
}
__device__ __forceinline__ float bf2f(unsigned short h) {
    return __uint_as_float(((unsigned int)h) << 16);
}

union FragU { v16bf v; v4u q[2]; };

// ===================================================================
// Kernel 1: wqsum[i] = sum_o Wq[i][o]
// ===================================================================
__global__ __launch_bounds__(256)
void wq_rowsum_kernel(const float* __restrict__ Wq, float* __restrict__ wqsum) {
    __shared__ float part[256];
    int tid  = threadIdx.x;
    int wave = tid >> 5, lane = tid & 31;
    int row  = blockIdx.x * 8 + wave;
    const float* p = Wq + (size_t)row * OO;
    float s = 0.f;
    for (int j = lane; j < OO; j += 32) s += p[j];
    part[tid] = s;
    __syncthreads();
    if (tid < 8) {
        float t = 0.f;
        #pragma unroll
        for (int j = 0; j < 32; ++j) t += part[tid * 32 + j];
        wqsum[blockIdx.x * 8 + tid] = t;
    }
}

// ===================================================================
// Kernel 2: softmax attention rows for the 36 unmasked (n,l) pairs
// ===================================================================
__global__ __launch_bounds__(256)
void attn_rows_kernel(const float* __restrict__ X, const float* __restrict__ Y,
                      const float* __restrict__ Wk, const float* __restrict__ wqsum,
                      float* __restrict__ attn) {
    __shared__ __align__(16) float yrow[II];
    __shared__ float red[256];
    int idx = blockIdx.x;
    int n = 0, base = 0;
    while (idx >= base + n + 1) { base += n + 1; ++n; }
    int l   = idx - base;
    int row = n * LL + l;
    int tid = threadIdx.x;

    ((float4*)yrow)[tid] = ((const float4*)(Y + (size_t)row * II))[tid];
    const float* xp = X + (size_t)row * II;
    float sp = 0.f;
    for (int j = tid; j < II; j += 256) sp += xp[j] * wqsum[j];
    red[tid] = sp;
    __syncthreads();
    for (int off = 128; off > 0; off >>= 1) {
        if (tid < off) red[tid] += red[tid + off];
        __syncthreads();
    }
    float sval = red[0] * (1.0f / 32.0f);   // / sqrt(O)
    __syncthreads();

    float acc0 = 0.f, acc1 = 0.f, acc2 = 0.f, acc3 = 0.f;
    for (int i = 0; i < II; ++i) {
        float yv = yrow[i];
        const float* wr = Wk + (size_t)i * OO;
        acc0 += yv * wr[tid];
        acc1 += yv * wr[tid + 256];
        acc2 += yv * wr[tid + 512];
        acc3 += yv * wr[tid + 768];
    }
    float sc0 = acc0 * sval, sc1 = acc1 * sval, sc2 = acc2 * sval, sc3 = acc3 * sval;

    float m = fmaxf(fmaxf(sc0, sc1), fmaxf(sc2, sc3));
    red[tid] = m;
    __syncthreads();
    for (int off = 128; off > 0; off >>= 1) {
        if (tid < off) red[tid] = fmaxf(red[tid], red[tid + off]);
        __syncthreads();
    }
    float rmax = red[0];
    __syncthreads();
    float e0 = __expf(sc0 - rmax), e1 = __expf(sc1 - rmax);
    float e2 = __expf(sc2 - rmax), e3 = __expf(sc3 - rmax);
    red[tid] = (e0 + e1) + (e2 + e3);
    __syncthreads();
    for (int off = 128; off > 0; off >>= 1) {
        if (tid < off) red[tid] += red[tid + off];
        __syncthreads();
    }
    float inv = 1.0f / red[0];
    float* arow = attn + (size_t)idx * OO;
    arow[tid]       = e0 * inv;
    arow[tid + 256] = e1 * inv;
    arow[tid + 512] = e2 * inv;
    arow[tid + 768] = e3 * inv;
}

// ===================================================================
// Kernel 3: split Y (fp32) into bf16 hi/lo planes (row-major)
// ===================================================================
__global__ __launch_bounds__(256)
void split_convert_kernel(const float* __restrict__ src,
                          unsigned short* __restrict__ hi,
                          unsigned short* __restrict__ lo, int n4) {
    for (int i = blockIdx.x * 256 + threadIdx.x; i < n4; i += gridDim.x * 256) {
        float4 a = ((const float4*)src)[i];
        unsigned short h0 = f2bf(a.x), h1 = f2bf(a.y), h2 = f2bf(a.z), h3 = f2bf(a.w);
        unsigned short g0 = f2bf(a.x - bf2f(h0)), g1 = f2bf(a.y - bf2f(h1));
        unsigned short g2 = f2bf(a.z - bf2f(h2)), g3 = f2bf(a.w - bf2f(h3));
        uint2 uh, ul;
        uh.x = (unsigned)h0 | ((unsigned)h1 << 16);
        uh.y = (unsigned)h2 | ((unsigned)h3 << 16);
        ul.x = (unsigned)g0 | ((unsigned)g1 << 16);
        ul.y = (unsigned)g2 | ((unsigned)g3 << 16);
        ((uint2*)hi)[i] = uh;
        ((uint2*)lo)[i] = ul;
    }
}

// ===================================================================
// Kernel 4: transpose + split Wv -> Wvt_hi/Wvt_lo, layout [n][k] bf16
// ===================================================================
__global__ __launch_bounds__(256)
void wv_transpose_split_kernel(const float* __restrict__ Wv,
                               unsigned short* __restrict__ thi,
                               unsigned short* __restrict__ tlo) {
    __shared__ __align__(16) float tile[64][68];
    int tid = threadIdx.x;
    int k0 = blockIdx.y * 64, n0 = blockIdx.x * 64;
    #pragma unroll
    for (int it = 0; it < 4; ++it) {
        int slot = tid + it * 256;
        int r  = slot >> 4;
        int c4 = (slot & 15) << 2;
        float4 v = *(const float4*)(Wv + (size_t)(k0 + r) * OO + n0 + c4);
        *(float4*)&tile[r][c4] = v;
    }
    __syncthreads();
    #pragma unroll
    for (int it = 0; it < 4; ++it) {
        int slot = tid + it * 256;
        int nn = slot >> 4;
        int k4 = (slot & 15) << 2;
        uint2 uh, ul;
        unsigned short h, g;
        float x;
        x = tile[k4 + 0][nn]; h = f2bf(x); g = f2bf(x - bf2f(h));
        uh.x = (unsigned)h;        ul.x = (unsigned)g;
        x = tile[k4 + 1][nn]; h = f2bf(x); g = f2bf(x - bf2f(h));
        uh.x |= (unsigned)h << 16; ul.x |= (unsigned)g << 16;
        x = tile[k4 + 2][nn]; h = f2bf(x); g = f2bf(x - bf2f(h));
        uh.y = (unsigned)h;        ul.y = (unsigned)g;
        x = tile[k4 + 3][nn]; h = f2bf(x); g = f2bf(x - bf2f(h));
        uh.y |= (unsigned)h << 16; ul.y |= (unsigned)g << 16;
        size_t o = (size_t)(n0 + nn) * II + k0 + k4;
        *(uint2*)&thi[o] = uh;
        *(uint2*)&tlo[o] = ul;
    }
}

// ===================================================================
// Kernel 5: out = (Y @ Wv) * attn_factor  — bf16x3 WMMA GEMM with
//   double-buffered LDS fed by GLOBAL_LOAD_ASYNC_TO_LDS_B128
// ===================================================================
#define BM 128
#define BN 128
#define KB 32
#define LDT 40                       // LDS row stride in bf16 (80 B)
#define BUFB (BM * LDT * 2)          // 10240 B per plane per buffer

struct AsyncSlot { unsigned lds; const char* g; };

__device__ __forceinline__ void issue_async8(const AsyncSlot* as, int buf, int kb) {
    size_t kb2  = (size_t)kb * 2;
    unsigned bo = buf ? (unsigned)BUFB : 0u;
    #pragma unroll
    for (int j = 0; j < 8; ++j) {
        asm volatile("global_load_async_to_lds_b128 %0, %1, off"
                     :: "v"(as[j].lds + bo), "v"(as[j].g + kb2)
                     : "memory");
    }
}

__global__ __launch_bounds__(256)
void gemm_attn_kernel(const unsigned short* __restrict__ Yhi,
                      const unsigned short* __restrict__ Ylo,
                      const unsigned short* __restrict__ Bthi,
                      const unsigned short* __restrict__ Btlo,
                      const float* __restrict__ attn,
                      float* __restrict__ out) {
    __shared__ __align__(16) unsigned short Ahi[2][BM * LDT];
    __shared__ __align__(16) unsigned short Alo[2][BM * LDT];
    __shared__ __align__(16) unsigned short Bhi[2][BN * LDT];
    __shared__ __align__(16) unsigned short Blo[2][BN * LDT];

    int tid   = threadIdx.x;
    int lane  = tid & 31, wave = tid >> 5;
    int half  = lane >> 4, lr = lane & 15;
    int waveM = wave >> 2, waveN = wave & 3;
    int bM    = blockIdx.y * BM;
    int bN    = blockIdx.x * BN;

    // ---- precompute per-thread async copy slots (8 x 16B chunks) ----
    unsigned aHiB = (unsigned)(unsigned long long)&Ahi[0][0];
    unsigned aLoB = (unsigned)(unsigned long long)&Alo[0][0];
    unsigned bHiB = (unsigned)(unsigned long long)&Bhi[0][0];
    unsigned bLoB = (unsigned)(unsigned long long)&Blo[0][0];

    AsyncSlot as[8];
    #pragma unroll
    for (int j = 0; j < 8; ++j) {
        int chunk = tid + j * 256;        // 0..2047
        int arr   = chunk >> 9;           // 0:Ahi 1:Alo 2:Bhi 3:Blo
        int w     = chunk & 511;
        int row   = w >> 2;               // 0..127
        int kc    = w & 3;                // 16B chunk within KB row
        unsigned ldsb; const char* gb;
        if      (arr == 0) { ldsb = aHiB; gb = (const char*)(Yhi  + (size_t)(bM + row) * II); }
        else if (arr == 1) { ldsb = aLoB; gb = (const char*)(Ylo  + (size_t)(bM + row) * II); }
        else if (arr == 2) { ldsb = bHiB; gb = (const char*)(Bthi + (size_t)(bN + row) * II); }
        else               { ldsb = bLoB; gb = (const char*)(Btlo + (size_t)(bN + row) * II); }
        as[j].lds = ldsb + (unsigned)(row * (LDT * 2) + kc * 16);
        as[j].g   = gb + kc * 16;
    }

    v8f c[4][2] = {};

    const int S = II / KB;                // 32 stages
    issue_async8(as, 0, 0);               // prologue: stage 0 -> buf 0

    for (int s = 0; s < S; ++s) {
        int buf = s & 1;
        if (s + 1 < S) {
            issue_async8(as, (s + 1) & 1, (s + 1) * KB);
            asm volatile("s_wait_asynccnt 8" ::: "memory");   // stage s landed
        } else {
            asm volatile("s_wait_asynccnt 0" ::: "memory");
        }
        __syncthreads();

        const unsigned short* Ah = Ahi[buf];
        const unsigned short* Al = Alo[buf];
        const unsigned short* Bh = Bhi[buf];
        const unsigned short* Bl = Blo[buf];

        // B fragments: col N = lr, K = half*8 .. +7 and 16 + half*8 .. +7
        FragU bh[2], bl[2];
        #pragma unroll
        for (int nt = 0; nt < 2; ++nt) {
            int bb = (waveN * 32 + nt * 16 + lr) * LDT + half * 8;
            bh[nt].q[0] = *(const v4u*)&Bh[bb];
            bh[nt].q[1] = *(const v4u*)&Bh[bb + 16];
            bl[nt].q[0] = *(const v4u*)&Bl[bb];
            bl[nt].q[1] = *(const v4u*)&Bl[bb + 16];
        }
        // A fragments + 3x WMMA (hi*hi + hi*lo + lo*hi)
        #pragma unroll
        for (int mt = 0; mt < 4; ++mt) {
            FragU ah, al;
            int ab = (waveM * 64 + mt * 16 + lr) * LDT + half * 8;
            ah.q[0] = *(const v4u*)&Ah[ab];
            ah.q[1] = *(const v4u*)&Ah[ab + 16];
            al.q[0] = *(const v4u*)&Al[ab];
            al.q[1] = *(const v4u*)&Al[ab + 16];
            #pragma unroll
            for (int nt = 0; nt < 2; ++nt) {
                c[mt][nt] = __builtin_amdgcn_wmma_f32_16x16x32_bf16(
                    false, ah.v, false, bh[nt].v, (short)0, c[mt][nt], false, false);
                c[mt][nt] = __builtin_amdgcn_wmma_f32_16x16x32_bf16(
                    false, ah.v, false, bl[nt].v, (short)0, c[mt][nt], false, false);
                c[mt][nt] = __builtin_amdgcn_wmma_f32_16x16x32_bf16(
                    false, al.v, false, bh[nt].v, (short)0, c[mt][nt], false, false);
            }
        }
        __syncthreads();   // all waves done reading buf before it is re-filled
    }

    // ---- fused epilogue: out = v * attn_factor ----
    #pragma unroll
    for (int mt = 0; mt < 4; ++mt) {
        #pragma unroll
        for (int nt = 0; nt < 2; ++nt) {
            int col = bN + waveN * 32 + nt * 16 + lr;
            #pragma unroll
            for (int r = 0; r < 8; ++r) {
                int row = bM + waveM * 64 + mt * 16 + half * 8 + r;
                float v = c[mt][nt][r];
                int sn = row >> 11;
                int sl = row & (LL - 1);
                float o;
                if (sl > sn) {
                    o = v * (1.0f / 1024.0f);
                } else {
                    size_t aidx = (size_t)(((sn * (sn + 1)) >> 1) + sl) * OO + col;
                    o = v * attn[aidx];
                }
                out[(size_t)row * OO + col] = o;
            }
        }
    }
}

// ===================================================================
extern "C" void kernel_launch(void* const* d_in, const int* in_sizes, int n_in,
                              void* d_out, int out_size, void* d_ws, size_t ws_size,
                              hipStream_t stream) {
    (void)in_sizes; (void)n_in; (void)out_size; (void)ws_size;
    const float* X  = (const float*)d_in[0];
    const float* Y  = (const float*)d_in[1];
    const float* Wq = (const float*)d_in[2];
    const float* Wk = (const float*)d_in[3];
    const float* Wv = (const float*)d_in[4];
    float* out = (float*)d_out;

    // workspace layout
    char* ws = (char*)d_ws;
    float* wqsum = (float*)ws;                                  // 4 KB
    float* attn  = (float*)(ws + (4 << 10));                    // 144 KB
    unsigned short* Yhi   = (unsigned short*)(ws + (256 << 10));
    unsigned short* Ylo   = Yhi   + (size_t)NROWS * II;
    unsigned short* Wvthi = Ylo   + (size_t)NROWS * II;
    unsigned short* Wvtlo = Wvthi + (size_t)II * OO;

    wq_rowsum_kernel<<<dim3(128), dim3(256), 0, stream>>>(Wq, wqsum);
    attn_rows_kernel<<<dim3(36), dim3(256), 0, stream>>>(X, Y, Wk, wqsum, attn);
    split_convert_kernel<<<dim3(4096), dim3(256), 0, stream>>>(Y, Yhi, Ylo,
                                                               (NROWS * II) / 4);
    wv_transpose_split_kernel<<<dim3(16, 16), dim3(256), 0, stream>>>(Wv, Wvthi, Wvtlo);
    gemm_attn_kernel<<<dim3(OO / BN, NROWS / BM), dim3(256), 0, stream>>>(
        Yhi, Ylo, Wvthi, Wvtlo, attn, out);
}
